// GRU_1108101562911
// MI455X (gfx1250) — matrix operand order repaired and checked by reading, not compiled, not run on hardware
//
#include <hip/hip_runtime.h>
#include <math.h>

// Problem dims (fixed by the reference)
#define BB 128
#define TT 1024
#define DD 256
#define UU 256
// Gate GEMM: A=[x_t | h] (16 x 512) @ [W_zr; U_zr] (512 x 512)
// Cand GEMM: A=[x_t | R*h] (16 x 512) @ [W_h ; U_h ] (512 x 256)
#define KTOT 512
#define N1 512
#define N2 256

typedef __attribute__((ext_vector_type(16))) __bf16 v16bf;
typedef __attribute__((ext_vector_type(8)))  float  v8f;

// ---- fast transcendental helpers (v_exp_f32 + v_rcp_f32, no IEEE div) -----
static __device__ __forceinline__ float fast_rcp(float x) {
#if __has_builtin(__builtin_amdgcn_rcpf)
    return __builtin_amdgcn_rcpf(x);
#else
    return 1.0f / x;
#endif
}

static __device__ __forceinline__ float fast_sigmoid(float x) {
    return fast_rcp(1.0f + __expf(-x));
}

static __device__ __forceinline__ float fast_tanh(float x) {
#if __has_builtin(__builtin_amdgcn_tanhf)
    return __builtin_amdgcn_tanhf(x);
#else
    const float t = __expf(-2.0f * x);
    return (1.0f - t) * fast_rcp(1.0f + t);
#endif
}

// ---------------------------------------------------------------------------
// Weight repack: fp32 [K x N] row-major (K-concatenation of two matrices)
// -> bf16 WMMA-B fragment order.
// B-matrix 32x16 bf16 layout (wave32): lane L holds column N = L%16;
// dword v, slot s holds K = (v<4?0:16) + (L/16)*8 + 2*(v&3) + s within the 32-K tile.
// Packed order: ((kt*Ntiles + nt)*32 + lane)*8 dwords, contiguous per lane.
// ---------------------------------------------------------------------------
__global__ __launch_bounds__(256) void repack_b(const float* __restrict__ top,
                                                const float* __restrict__ bot,
                                                __bf16* __restrict__ out,
                                                int N /*columns*/)
{
    const int d    = blockIdx.x * 256 + threadIdx.x;   // dword id
    const int v    = d & 7;
    const int lane = (d >> 3) & 31;
    const int tf   = d >> 8;                           // flat tile index
    const int ntiles = N >> 4;
    const int nt   = tf % ntiles;
    const int kt   = tf / ntiles;
    const int K0   = kt * 32 + ((v >> 2) << 4) + ((lane >> 4) << 3) + 2 * (v & 3);
    const int col  = nt * 16 + (lane & 15);
    const float f0 = (K0     < DD) ? top[(size_t)K0 * N + col]
                                   : bot[(size_t)(K0 - DD) * N + col];
    const float f1 = (K0 + 1 < DD) ? top[(size_t)(K0 + 1) * N + col]
                                   : bot[(size_t)(K0 + 1 - DD) * N + col];
    out[2 * (size_t)d]     = (__bf16)f0;
    out[2 * (size_t)d + 1] = (__bf16)f1;
}

// ---------------------------------------------------------------------------
// Persistent GRU scan. The recurrence is independent per batch row, so each
// block owns rows [r0, r0+16) and runs all 1024 timesteps with h resident in
// LDS. 512 threads = 16 waves: wave w owns gate N-tiles {2w,2w+1} (cols
// 32w..32w+31 of 512) and candidate N-tile w (cols 16w..16w+15 of 256).
// ---------------------------------------------------------------------------
__global__ __launch_bounds__(512) void gru_scan(
    const float*  __restrict__ x,     // [128,1024,256] fp32
    const __bf16* __restrict__ pG1,   // packed [W_zr; U_zr]  512x512
    const __bf16* __restrict__ pG2,   // packed [W_h ; U_h ]  512x256
    const float*  __restrict__ bzr,   // [512]
    const float*  __restrict__ bh,    // [256]
    float*        __restrict__ out)   // [128,256] fp32
{
    __shared__ float hL[16 * UU];     // resident hidden state (fp32)
    __shared__ float ldsG[16 * N1];   // gates: cols 0..255 = Z, 256..511 = R

    const int tid  = threadIdx.x;
    const int wave = tid >> 5;
    const int lane = tid & 31;
    const int hl   = lane >> 4;       // lane half (0/1)
    const int lcol = lane & 15;       // N within tile / A row M
    const int r0   = blockIdx.x * 16;

    // h0 = 0
    for (int i = tid; i < 16 * UU; i += 512) hL[i] = 0.0f;
    __syncthreads();

    // Hoist biases (constant over t)
    const float bz0 = bzr[(wave * 2 + 0) * 16 + lcol];
    const float bz1 = bzr[(wave * 2 + 1) * 16 + lcol];
    const float bh0 = bh[wave * 16 + lcol];

    const float* xbase = x + (size_t)(r0 + lcol) * TT * DD;  // this lane's A-row
    const float* hrow  = &hL[lcol * UU];
    const float* rrow  = &ldsG[lcol * N1 + UU];               // reset gate row

    const v8f zf = {0.f, 0.f, 0.f, 0.f, 0.f, 0.f, 0.f, 0.f};

    // Opaque zero offset (in elements), redefined every iteration: prevents
    // LICM from hoisting the 48 loop-invariant B-fragment loads (384 VGPRs
    // -> scratch spills) while keeping pG1/pG2 as known-global pointers so
    // the loads stay global_load_b128 (LOADcnt only), not flat (LOADcnt +
    // DScnt, arbitrating against the LDS pipe we use for h/gates).
    size_t woff = 0;

    for (int t = 0; t < TT; ++t) {
        asm volatile("" : "+s"(woff));   // opaque, always 0, zero cost

        const float* xrow = xbase + (size_t)t * DD;
        if (t + 1 < TT) {
            __builtin_prefetch(xrow + DD + hl * 128, 0, 1);   // next timestep
        }

        // ------------- GEMM1: gates = [x_t | h] @ [W_zr; U_zr] -------------
        // A fragment (16x32 bf16): lane holds row M=lane%16;
        // dword v slot s: K = (v<4?0:16) + (lane/16)*8 + 2*(v&3) + s.
        v8f acc0 = zf, acc1 = zf;
        v16bf xa[8];                   // cached x-side fragments for GEMM2
        #pragma unroll
        for (int c = 0; c < 16; ++c) {
            const int kb = (c & 7) * 32;
            v16bf a;
            if (c < 8) {
                #pragma unroll
                for (int v = 0; v < 8; ++v) {
                    const int kk = kb + ((v < 4) ? 0 : 16) + hl * 8 + 2 * (v & 3);
                    a[2 * v]     = (__bf16)xrow[kk];
                    a[2 * v + 1] = (__bf16)xrow[kk + 1];
                }
                xa[c] = a;
            } else {                   // recurrent half: h from LDS
                #pragma unroll
                for (int v = 0; v < 8; ++v) {
                    const int kk = kb + ((v < 4) ? 0 : 16) + hl * 8 + 2 * (v & 3);
                    a[2 * v]     = (__bf16)hrow[kk];
                    a[2 * v + 1] = (__bf16)hrow[kk + 1];
                }
            }
            const v16bf b0 = *(const v16bf*)(pG1 + woff + ((size_t)(c * (N1 >> 4) + wave * 2 + 0) * 32 + lane) * 16);
            const v16bf b1 = *(const v16bf*)(pG1 + woff + ((size_t)(c * (N1 >> 4) + wave * 2 + 1) * 32 + lane) * 16);
            acc0 = __builtin_amdgcn_wmma_f32_16x16x32_bf16(false, a, false, b0, (short)0, acc0, false, false);
            acc1 = __builtin_amdgcn_wmma_f32_16x16x32_bf16(false, a, false, b1, (short)0, acc1, false, false);
        }

        // Epilogue 1: bias + sigmoid -> LDS gates (branch-free).
        // C layout: VGPR i of lane L holds (M = 8*(L/16)+i, N = L%16).
        #pragma unroll
        for (int i = 0; i < 8; ++i) {
            const int mrow = hl * 8 + i;
            ldsG[mrow * N1 + (wave * 2 + 0) * 16 + lcol] = fast_sigmoid(acc0[i] + bz0);
            ldsG[mrow * N1 + (wave * 2 + 1) * 16 + lcol] = fast_sigmoid(acc1[i] + bz1);
        }
        __syncthreads();               // gates visible to all waves

        // ------------- GEMM2: cand = [x_t | R*h] @ [W_h; U_h] --------------
        v8f acc2 = zf;
        #pragma unroll
        for (int c = 0; c < 16; ++c) {
            v16bf a;
            if (c < 8) {
                a = xa[c];
            } else {
                const int kb = (c & 7) * 32;
                #pragma unroll
                for (int v = 0; v < 8; ++v) {
                    const int kk = kb + ((v < 4) ? 0 : 16) + hl * 8 + 2 * (v & 3);
                    a[2 * v]     = (__bf16)(rrow[kk]     * hrow[kk]);
                    a[2 * v + 1] = (__bf16)(rrow[kk + 1] * hrow[kk + 1]);
                }
            }
            const v16bf b = *(const v16bf*)(pG2 + woff + ((size_t)(c * (N2 >> 4) + wave) * 32 + lane) * 16);
            acc2 = __builtin_amdgcn_wmma_f32_16x16x32_bf16(false, a, false, b, (short)0, acc2, false, false);
        }
        __syncthreads();               // all reads of h done before update

        // Epilogue 2: tanh + blend, update h in LDS (and out on last step).
        {
            const int col  = wave * 16 + lcol;
            const bool last = (t == TT - 1);   // uniform
            #pragma unroll
            for (int i = 0; i < 8; ++i) {
                const int mrow = hl * 8 + i;
                const float S  = fast_tanh(acc2[i] + bh0);
                const float z  = ldsG[mrow * N1 + col];           // Z
                const float hv = hL[mrow * UU + col];
                const float hn = (1.0f - z) * hv + z * S;
                hL[mrow * UU + col] = hn;
                if (last) out[(size_t)(r0 + mrow) * UU + col] = hn;
            }
        }
        __syncthreads();               // h update visible before next GEMM1
    }
}

// ---------------------------------------------------------------------------
extern "C" void kernel_launch(void* const* d_in, const int* in_sizes, int n_in,
                              void* d_out, int out_size, void* d_ws, size_t ws_size,
                              hipStream_t stream)
{
    (void)in_sizes; (void)n_in; (void)out_size; (void)ws_size;
    const float* x   = (const float*)d_in[0];
    const float* Wzr = (const float*)d_in[1];
    const float* Uzr = (const float*)d_in[2];
    const float* bzr = (const float*)d_in[3];
    const float* Wh  = (const float*)d_in[4];
    const float* Uh  = (const float*)d_in[5];
    const float* bh  = (const float*)d_in[6];
    float* out = (float*)d_out;

    // Workspace: packed bf16 weights only (hot in L2 for the whole scan).
    char* ws = (char*)d_ws;
    __bf16* pG1 = (__bf16*)ws;                            // 512*512 bf16 = 512 KB
    __bf16* pG2 = (__bf16*)(ws + (size_t)512 * 1024);     // 256 KB

    repack_b<<<(KTOT * N1 / 2) / 256, 256, 0, stream>>>(Wzr, Uzr, pG1, N1); // 512 blocks
    repack_b<<<(KTOT * N2 / 2) / 256, 256, 0, stream>>>(Wh,  Uh,  pG2, N2); // 256 blocks

    // Persistent scan: batch rows are independent -> no inter-block sync,
    // no per-step launches. 8 blocks x 512 threads, h resident in LDS.
    gru_scan<<<BB / 16, 512, 0, stream>>>(x, pG1, pG2, bzr, bh, out);
}
